// CoLoRConv2d_81277961109972
// MI455X (gfx1250) — matrix operand order repaired and checked by reading, not compiled, not run on hardware
//
#include <hip/hip_runtime.h>
#include <hip/hip_bf16.h>

typedef __attribute__((ext_vector_type(16))) _Float16 v16h;
typedef __attribute__((ext_vector_type(8)))  _Float16 v8h;
typedef __attribute__((ext_vector_type(8)))  float    v8f;
typedef __attribute__((ext_vector_type(4)))  unsigned int v4u;

// Problem constants
#define C_IN   512
#define C_OUT  512
#define HSZ    64
#define WSZ    64
#define BATCH  8
#define RANK   4
#define NTAPS  9
#define KCH    32              // input-channel chunk per WMMA K
#define NKC    (C_IN / KCH)    // 16 K-chunks
#define NOB    (C_OUT / 16)    // 32 o-blocks of 16
#define MOD_SCALE   0.044194173824159216f  /* 1/sqrt(512) */
#define CONV_SCALE2 (1.0f / 4608.0f)       /* (1/sqrt(512*9))^2 */

// ---------------- workspace layout (bytes) ----------------
static constexpr size_t OFF_W5    = 0;                          // [512][512][9] f32
static constexpr size_t SZ_W5     = (size_t)C_OUT * C_IN * NTAPS * 4;
static constexpr size_t OFF_Q     = OFF_W5 + SZ_W5;             // [512][512] f32
static constexpr size_t SZ_Q      = (size_t)C_OUT * C_IN * 4;
static constexpr size_t OFF_WPACK = OFF_Q + SZ_Q;               // [9*32*16][512] f16 (A-fragments)
static constexpr size_t SZ_WPACK  = (size_t)NTAPS * NOB * NKC * 512 * 2;
static constexpr size_t OFF_S     = OFF_WPACK + SZ_WPACK;       // [8][512] f32
static constexpr size_t OFF_S2    = OFF_S  + (size_t)BATCH * C_IN * 4;
static constexpr size_t OFF_DEMOD = OFF_S2 + (size_t)BATCH * C_IN * 4;
static constexpr size_t OFF_XT    = OFF_DEMOD + (size_t)BATCH * C_OUT * 4; // [8][16][64][64][32] f16

// ================= kernel 1: LoRA conv-weight generation =================
__global__ __launch_bounds__(256) void k_wgen(const float* __restrict__ weight,
                                              const float* __restrict__ llB,
                                              const float* __restrict__ llInst,
                                              const float* __restrict__ llA,
                                              float* __restrict__ w5,
                                              float* __restrict__ q) {
  const int o = blockIdx.x;
  float Bo[RANK];
#pragma unroll
  for (int r = 0; r < RANK; ++r) Bo[r] = llB[o * RANK + r];
  float Bm[RANK][NTAPS];
#pragma unroll
  for (int rb = 0; rb < RANK; ++rb)
#pragma unroll
    for (int t = 0; t < NTAPS; ++t) {
      float acc = 0.f;
#pragma unroll
      for (int r = 0; r < RANK; ++r) acc += Bo[r] * llInst[(rb * RANK + r) * NTAPS + t];
      Bm[rb][t] = fmaxf(acc, 0.f);
    }
  for (int i = threadIdx.x; i < C_IN; i += 256) {
    float Ai[RANK];
#pragma unroll
    for (int r = 0; r < RANK; ++r) Ai[r] = llA[i * RANK + r];
    float qs = 0.f;
#pragma unroll
    for (int t = 0; t < NTAPS; ++t) {
      float wa = 0.f;
#pragma unroll
      for (int rb = 0; rb < RANK; ++rb) wa += Ai[rb] * Bm[rb][t];
      wa = fmaxf(wa, 0.f);
      float w = weight[((size_t)o * C_IN + i) * NTAPS + t] + wa;
      w5[((size_t)o * C_IN + i) * NTAPS + t] = w;
      qs += w * w;
    }
    q[(size_t)o * C_IN + i] = qs;
  }
}

// ================= kernel 2: style modulation s[b,i] =================
__global__ __launch_bounds__(256) void k_style(const float* __restrict__ style,
                                               const float* __restrict__ mw,
                                               const float* __restrict__ mb,
                                               const float* __restrict__ fcA,
                                               const float* __restrict__ fcB,
                                               const float* __restrict__ fcBias,
                                               float* __restrict__ s,
                                               float* __restrict__ s2) {
  const int b = blockIdx.x;
  __shared__ float sty[C_IN];
  for (int j = threadIdx.x; j < C_IN; j += 256) sty[j] = style[(size_t)b * C_IN + j];
  __syncthreads();
  float sa[RANK] = {0.f, 0.f, 0.f, 0.f};
  for (int j = 0; j < C_IN; ++j) {
    float sv = sty[j];
#pragma unroll
    for (int r = 0; r < RANK; ++r) sa[r] += sv * fcA[j * RANK + r];
  }
  for (int i = threadIdx.x; i < C_IN; i += 256) {
    float dot = 0.f;
    for (int j = 0; j < C_IN; ++j) dot += sty[j] * mw[(size_t)i * C_IN + j];
    float lor = 0.f;
#pragma unroll
    for (int r = 0; r < RANK; ++r) lor += fcB[i * RANK + r] * sa[r];
    float val = MOD_SCALE * (dot + lor) + mb[i] + fcBias[i];
    s[(size_t)b * C_IN + i] = val;
    s2[(size_t)b * C_IN + i] = val * val;
  }
}

// ================= kernel 3: demod[b,o] =================
__global__ __launch_bounds__(256) void k_demod(const float* __restrict__ s2,
                                               const float* __restrict__ q,
                                               float* __restrict__ demod) {
  const int b = blockIdx.x;
  for (int o = threadIdx.x; o < C_OUT; o += 256) {
    float acc = 0.f;
    for (int i = 0; i < C_IN; ++i) acc += s2[(size_t)b * C_IN + i] * q[(size_t)o * C_IN + i];
    demod[(size_t)b * C_OUT + o] = rsqrtf(CONV_SCALE2 * acc + 1e-8f);
  }
}

// ================= kernel 4: pack w5 into WMMA A-fragment layout =================
// frag = (t*NOB + ob)*NKC + kc ; lane l stores its 16 halfs contiguously.
// A 16x32 packing used here (contiguous k, same as B): row M = l%16 ; h = l>>4 ; K=16h+e.
__global__ __launch_bounds__(256) void k_pack(const float* __restrict__ w5,
                                              _Float16* __restrict__ wpack) {
  const int lane = threadIdx.x & 31;
  const int frag = blockIdx.x * 8 + (threadIdx.x >> 5);
  const int t  = frag / (NOB * NKC);
  const int ob = (frag / NKC) % NOB;
  const int kc = frag % NKC;
  const int h = lane >> 4, l16 = lane & 15;
  const int o = ob * 16 + l16;
  _Float16* dst = wpack + (size_t)frag * 512 + lane * 16;
#pragma unroll
  for (int e = 0; e < 16; ++e) {
    int K = 16 * h + e;
    int i = kc * KCH + K;
    dst[e] = (_Float16)w5[((size_t)o * C_IN + i) * NTAPS + t];
  }
}

// ================= kernel 5: x' = s*x, f16, channel-last tiled layout =================
__global__ __launch_bounds__(256) void k_premod(const float* __restrict__ input,
                                                const float* __restrict__ s,
                                                _Float16* __restrict__ xT) {
  const int hrow = blockIdx.x, kc = blockIdx.y, b = blockIdx.z;
  __shared__ _Float16 lp[KCH][WSZ + 2];
  const int w = threadIdx.x & 63, csub = threadIdx.x >> 6;
#pragma unroll
  for (int it = 0; it < 8; ++it) {
    int c = it * 4 + csub;
    int i = kc * KCH + c;
    float v = input[(((size_t)b * C_IN + i) * HSZ + hrow) * WSZ + w] * s[(size_t)b * C_IN + i];
    lp[c][w] = (_Float16)v;
  }
  __syncthreads();
  const int c2 = threadIdx.x & 31, wq = threadIdx.x >> 5;
  const size_t base = ((((size_t)b * NKC + kc) * HSZ + hrow) * WSZ) * KCH;
#pragma unroll
  for (int it = 0; it < 8; ++it) {
    int w2 = it * 8 + wq;
    xT[base + (size_t)w2 * KCH + c2] = lp[c2][w2];
  }
}

// ================= kernel 6: conv via WMMA + double-buffered async LDS staging =====
__global__ __launch_bounds__(256) void k_conv(const _Float16* __restrict__ xT,
                                              const _Float16* __restrict__ wpack,
                                              const float* __restrict__ demod,
                                              float* __restrict__ out) {
  const int y = blockIdx.x, ot = blockIdx.y, b = blockIdx.z;
  const int tid = threadIdx.x;
  const int wv = tid >> 5, lane = tid & 31;
  const int h = lane >> 4, l16 = lane & 15;
  const int m = wv & 3, np = wv >> 2;
  const int n0 = np * 2, n1 = n0 + 1;
  const int ob = ot * 4 + m;

  // double-buffered: [buf][row 0..2][col 0..65 halo][chan padded to 40]
  __shared__ alignas(16) _Float16 ldsX[2][3][66][40];

  // ---- one-time zero fill (halo cols, OOB rows, padding are kc-invariant) ----
  {
    v4u z = {0u, 0u, 0u, 0u};
    char* base = (char*)&ldsX[0][0][0][0];
    for (int c = tid; c < (int)(sizeof(ldsX) / 16); c += 256)
      *(v4u*)(base + (size_t)c * 16) = z;
  }
  __syncthreads();

  // per-block global base for x' chunks
  const size_t xbase = (((size_t)b * NKC) * HSZ) * WSZ * KCH;
  const size_t kcstride = (size_t)HSZ * WSZ * KCH;

  // per-thread staging coordinates (3 chunks of 16B each)
  int sq4[3], scol[3], syy[3];
#pragma unroll
  for (int j = 0; j < 3; ++j) {
    int idx = tid + j * 256;        // (r:3) x (col:64) x (q:4)
    sq4[j]  = idx & 3;
    scol[j] = (idx >> 2) & 63;
    syy[j]  = y - 1 + (idx >> 8);
  }

  // issue async stage of k-chunk `kc` into buffer `buf`
  auto issue_stage = [&](int kc, int buf) {
#pragma unroll
    for (int j = 0; j < 3; ++j) {
      if (syy[j] >= 0 && syy[j] < HSZ) {
        int r = syy[j] - (y - 1);
        unsigned lds_off = (unsigned)(unsigned long long)
            (&ldsX[buf][r][scol[j] + 1][sq4[j] * 8]);
        unsigned long long ga = (unsigned long long)
            (xT + xbase + (size_t)kc * kcstride +
             ((size_t)syy[j] * WSZ + scol[j]) * KCH + sq4[j] * 8);
        asm volatile("global_load_async_to_lds_b128 %0, %1, off"
                     :: "v"(lds_off), "v"(ga) : "memory");
      }
    }
  };

  v8f acc0 = {};
  v8f acc1 = {};

  // A-fragment base for this wave (tap stride is a compile-time 512KB offset)
  const _Float16* aw = wpack + (size_t)ob * NKC * 512 + (size_t)lane * 16;

  issue_stage(0, 0);
  asm volatile("s_wait_asynccnt 0x0" ::: "memory");
  __syncthreads();

  for (int kc = 0; kc < NKC; ++kc) {
    const int cur = kc & 1;
    if (kc + 1 < NKC) issue_stage(kc + 1, cur ^ 1);

    const _Float16* akc = aw + (size_t)kc * 512;
#pragma unroll
    for (int t = 0; t < NTAPS; ++t) {
      const int kh = t / 3, kw = t % 3;
      v16h a = *(const v16h*)(akc + (size_t)t * ((size_t)NOB * NKC * 512));

      const int cb0 = n0 * 16 + l16 + kw;
      v8h lo0 = *(const v8h*)&ldsX[cur][kh][cb0][16 * h];
      v8h hi0 = *(const v8h*)&ldsX[cur][kh][cb0][16 * h + 8];
      v16h bf0 = __builtin_shufflevector(lo0, hi0, 0, 1, 2, 3, 4, 5, 6, 7,
                                         8, 9, 10, 11, 12, 13, 14, 15);
      acc0 = __builtin_amdgcn_wmma_f32_16x16x32_f16(false, a, false, bf0,
                                                    (short)0, acc0, false, false);

      const int cb1 = n1 * 16 + l16 + kw;
      v8h lo1 = *(const v8h*)&ldsX[cur][kh][cb1][16 * h];
      v8h hi1 = *(const v8h*)&ldsX[cur][kh][cb1][16 * h + 8];
      v16h bf1 = __builtin_shufflevector(lo1, hi1, 0, 1, 2, 3, 4, 5, 6, 7,
                                         8, 9, 10, 11, 12, 13, 14, 15);
      acc1 = __builtin_amdgcn_wmma_f32_16x16x32_f16(false, a, false, bf1,
                                                    (short)0, acc1, false, false);
    }

    // async writes (next buf) complete + this wave's LDS reads done, then barrier
    asm volatile("s_wait_asynccnt 0x0" ::: "memory");
    __syncthreads();
  }

  // ---- epilogue: out = CONV_SCALE * demod[b,o] * acc ----
  const float conv_scale = rsqrtf(4608.0f);
  const int o0 = ot * 64 + m * 16 + 8 * h;
  const int x0 = n0 * 16 + l16, x1 = n1 * 16 + l16;
#pragma unroll
  for (int v = 0; v < 8; ++v) {
    int o = o0 + v;
    float sc = conv_scale * demod[(size_t)b * C_OUT + o];
    size_t ro = (((size_t)b * C_OUT + o) * HSZ + y) * WSZ;
    out[ro + x0] = acc0[v] * sc;
    out[ro + x1] = acc1[v] * sc;
  }
}

// ================= host launcher =================
extern "C" void kernel_launch(void* const* d_in, const int* in_sizes, int n_in,
                              void* d_out, int out_size, void* d_ws, size_t ws_size,
                              hipStream_t stream) {
  (void)in_sizes; (void)n_in; (void)out_size; (void)ws_size;
  const float* input    = (const float*)d_in[0];
  const float* style    = (const float*)d_in[1];
  const float* weight   = (const float*)d_in[2];
  const float* ll_B     = (const float*)d_in[3];
  const float* ll_Binst = (const float*)d_in[4];
  const float* ll_A     = (const float*)d_in[5];
  const float* mod_w    = (const float*)d_in[6];
  const float* mod_b    = (const float*)d_in[7];
  const float* fc_A     = (const float*)d_in[8];
  const float* fc_B     = (const float*)d_in[9];
  const float* fc_bias  = (const float*)d_in[10];
  float* out = (float*)d_out;

  char* ws = (char*)d_ws;
  float*    w5    = (float*)(ws + OFF_W5);
  float*    q     = (float*)(ws + OFF_Q);
  _Float16* wpack = (_Float16*)(ws + OFF_WPACK);
  float*    s     = (float*)(ws + OFF_S);
  float*    s2    = (float*)(ws + OFF_S2);
  float*    dem   = (float*)(ws + OFF_DEMOD);
  _Float16* xT    = (_Float16*)(ws + OFF_XT);

  k_wgen<<<C_OUT, 256, 0, stream>>>(weight, ll_B, ll_Binst, ll_A, w5, q);
  k_style<<<BATCH, 256, 0, stream>>>(style, mod_w, mod_b, fc_A, fc_B, fc_bias, s, s2);
  k_demod<<<BATCH, 256, 0, stream>>>(s2, q, dem);
  k_pack<<<(NTAPS * NOB * NKC) / 8, 256, 0, stream>>>(w5, wpack);
  k_premod<<<dim3(HSZ, NKC, BATCH), 256, 0, stream>>>(input, s, xT);
  k_conv<<<dim3(HSZ, C_OUT / 64, BATCH), 256, 0, stream>>>(xT, wpack, dem, out);
}
